// FatigueLSTM_36713380446334
// MI455X (gfx1250) — matrix operand order repaired and checked by reading, not compiled, not run on hardware
//
#include <hip/hip_runtime.h>

typedef _Float16 v16h __attribute__((ext_vector_type(16)));
typedef _Float16 v8h  __attribute__((ext_vector_type(8)));
typedef float    v8f  __attribute__((ext_vector_type(8)));

static constexpr int TT  = 512;  // timesteps
static constexpr int HID = 32;   // hidden size (== WMMA K)

__device__ __forceinline__ float fast_tanhf(float x) {
#if __has_builtin(__builtin_amdgcn_tanhf)
    return __builtin_amdgcn_tanhf(x);
#else
    float e = __builtin_amdgcn_exp2f(2.8853900817779268f * x);
    return 1.0f - 2.0f * __builtin_amdgcn_rcpf(e + 1.0f);
#endif
}

__device__ __forceinline__ float fast_sigmoidf(float x) {
#if __has_builtin(__builtin_amdgcn_tanhf)
    return __builtin_fmaf(0.5f, __builtin_amdgcn_tanhf(0.5f * x), 0.5f);
#else
    return __builtin_amdgcn_rcpf(1.0f + __builtin_amdgcn_exp2f(-1.4426950408889634f * x));
#endif
}

// One wave (32 lanes) owns a 16-row batch tile for all 512 timesteps.
// gates[16x128] = [x_t|1|0.. (K=6)] * Bx  +  h[16x32] * Bhh  via chained
// v_wmma_f32_16x16x32_f16 (8 gate tiles of 16 columns each).
// Bias (b_ih+b_hh) rides as K-row 5 of Bx against A[5]=1.0 so the x-side
// WMMA's C operand is the inline zero (no live bias-splat registers).
__global__ __launch_bounds__(32) void lstm_wmma_kernel(
    const float* __restrict__ x,     // [B, 512, 5]
    const float* __restrict__ W_ih,  // [128, 5]
    const float* __restrict__ W_hh,  // [128, 32]
    const float* __restrict__ b_ih,  // [128]
    const float* __restrict__ b_hh,  // [128]
    const float* __restrict__ W_fc,  // [3, 32]
    const float* __restrict__ b_fc,  // [3]
    float* __restrict__ out)         // [B, 3]
{
    const int lane = threadIdx.x & 31;
    const int n    = lane & 15;    // column within 16-wide tile / batch row for A-frag
    const int hi   = lane >> 4;    // lane half (selects K half of B / M half of D)
    const int row_base = blockIdx.x * 16;

    // 16 rows x 32 hidden, f16: h staging for D-layout -> A-fragment transpose.
    __shared__ __align__(16) _Float16 hbuf[16][32];

    // ---- W_hh as 8 WMMA B-fragments (32x16 f16 each) ----
    // B layout: col = lane&15; lanes 0-15 hold K=0..15, lanes 16-31 hold K=16..31.
    v16h bhh[8];
    {
        const int kbase = hi * 16;
        #pragma unroll
        for (int j = 0; j < 8; ++j) {
            const float* wr = W_hh + (j * 16 + n) * HID + kbase;  // row of W_hh, contiguous K
            #pragma unroll
            for (int k = 0; k < 16; ++k) bhh[j][k] = (_Float16)wr[k];
        }
    }

    // ---- W_ih + bias as B-fragments (K rows 0..4 = W_ih, row 5 = bias) ----
    v16h bx[8];
    #pragma unroll
    for (int j = 0; j < 8; ++j) {
        const int g = j * 16 + n;  // gate column index 0..127
        #pragma unroll
        for (int k = 0; k < 16; ++k) bx[j][k] = (_Float16)0.0f;
        if (hi == 0) {
            #pragma unroll
            for (int i = 0; i < 5; ++i) bx[j][i] = (_Float16)W_ih[g * 5 + i];
            bx[j][5] = (_Float16)(b_ih[g] + b_hh[g]);
        }
    }

    // ---- recurrent state in registers (D layout: 2 tiles of v8f) ----
    v8f c_lo, c_hi;
    #pragma unroll
    for (int e = 0; e < 8; ++e) { c_lo[e] = 0.0f; c_hi[e] = 0.0f; }
    v16h ah;  // h as A-fragment; h0 = 0
    #pragma unroll
    for (int e = 0; e < 16; ++e) ah[e] = (_Float16)0.0f;

    // A-fragment for the input GEMM, constant parts hoisted out of the loop:
    // K=5 = 1.0 (bias multiplier) on lanes 0-15; everything else starts 0 and
    // only elements 0..4 of lanes 0-15 are rewritten per timestep.
    v16h ax;
    #pragma unroll
    for (int e = 0; e < 16; ++e) ax[e] = (_Float16)0.0f;
    if (hi == 0) ax[5] = (_Float16)1.0f;

    // x prefetch (lanes 16-31 duplicate lane&15's loads; their A elems stay 0)
    const float* xp = x + (size_t)(row_base + n) * TT * 5;
    float xa[5];
    #pragma unroll
    for (int i = 0; i < 5; ++i) xa[i] = xp[i];

    for (int t = 0; t < TT; ++t) {
        // update only the x elements of the A-fragment (lanes 0-15)
        if (hi == 0) {
            #pragma unroll
            for (int i = 0; i < 5; ++i) ax[i] = (_Float16)xa[i];
        }

        // prefetch x for t+1 (clamped at the end to stay in-bounds)
        const int tn = (t < TT - 1) ? (t + 1) : t;
        float xb[5];
        #pragma unroll
        for (int i = 0; i < 5; ++i) xb[i] = xp[tn * 5 + i];

        // gates = [x_t|1] * [W_ih^T; b]   (C = inline zero; independent of recurrence)
        v8f acc[8];
        {
            v8f zc;
            #pragma unroll
            for (int e = 0; e < 8; ++e) zc[e] = 0.0f;
            #pragma unroll
            for (int j = 0; j < 8; ++j) {
                acc[j] = __builtin_amdgcn_wmma_f32_16x16x32_f16(
                    false, ax, false, bx[j], (short)0, zc, false, false);
            }
        }
        // gates += h * W_hh^T  (the serial dependency)
        #pragma unroll
        for (int j = 0; j < 8; ++j) {
            acc[j] = __builtin_amdgcn_wmma_f32_16x16x32_f16(
                false, ah, false, bhh[j], (short)0, acc[j], false, false);
        }

        // Elementwise LSTM cell. Tiles: i=0/1, f=2/3, g=4/5, o=6/7 (lo=hidden 0-15).
        #pragma unroll
        for (int v = 0; v < 8; ++v) {
            float i0 = fast_sigmoidf(acc[0][v]);
            float f0 = fast_sigmoidf(acc[2][v]);
            float g0 = fast_tanhf   (acc[4][v]);
            float o0 = fast_sigmoidf(acc[6][v]);
            c_lo[v]  = __builtin_fmaf(f0, c_lo[v], i0 * g0);
            float h0 = o0 * fast_tanhf(c_lo[v]);

            float i1 = fast_sigmoidf(acc[1][v]);
            float f1 = fast_sigmoidf(acc[3][v]);
            float g1 = fast_tanhf   (acc[5][v]);
            float o1 = fast_sigmoidf(acc[7][v]);
            c_hi[v]  = __builtin_fmaf(f1, c_hi[v], i1 * g1);
            float h1 = o1 * fast_tanhf(c_hi[v]);

            // D layout: element (m, n) with m = v + 8*hi, n = lane&15 (+16 for hi tile)
            const int m = v + hi * 8;
            hbuf[m][n]      = (_Float16)h0;
            hbuf[m][n + 16] = (_Float16)h1;
        }

        // Transpose h (via LDS) into the A-fragment layout for the next step:
        // lanes 0-15: K 0..15 of row n;  lanes 16-31: K 16..31.
        {
            const int k0 = hi * 16;
            v8h p0 = *(const v8h*)(&hbuf[n][k0]);
            v8h p1 = *(const v8h*)(&hbuf[n][k0 + 8]);
            #pragma unroll
            for (int e = 0; e < 8; ++e) { ah[e] = p0[e]; ah[e + 8] = p1[e]; }
        }

        #pragma unroll
        for (int i = 0; i < 5; ++i) xa[i] = xb[i];
    }

    // ---- final FC: out[m, 0..2] = h_last[m, :] @ W_fc^T + b_fc ----
    if (hi == 0) {
        float a0 = b_fc[0], a1 = b_fc[1], a2 = b_fc[2];
        #pragma unroll
        for (int k = 0; k < HID; ++k) {
            float hv = (float)hbuf[n][k];
            a0 = __builtin_fmaf(hv, W_fc[k],      a0);
            a1 = __builtin_fmaf(hv, W_fc[32 + k], a1);
            a2 = __builtin_fmaf(hv, W_fc[64 + k], a2);
        }
        float* o = out + (size_t)(row_base + n) * 3;
        o[0] = a0; o[1] = a1; o[2] = a2;
    }
}

extern "C" void kernel_launch(void* const* d_in, const int* in_sizes, int n_in,
                              void* d_out, int out_size, void* d_ws, size_t ws_size,
                              hipStream_t stream) {
    const float* x    = (const float*)d_in[0];
    const float* W_ih = (const float*)d_in[1];
    const float* W_hh = (const float*)d_in[2];
    const float* b_ih = (const float*)d_in[3];
    const float* b_hh = (const float*)d_in[4];
    const float* W_fc = (const float*)d_in[5];
    const float* b_fc = (const float*)d_in[6];
    float* out = (float*)d_out;

    const int B = in_sizes[0] / (TT * 5);   // 4096
    const int tiles = B / 16;               // 256 waves, one 32-thread block each

    lstm_wmma_kernel<<<dim3(tiles), dim3(32), 0, stream>>>(
        x, W_ih, W_hh, b_ih, b_hh, W_fc, b_fc, out);
}
